// PatchMoELayer_40733469835856
// MI455X (gfx1250) — matrix operand order repaired and checked by reading, not compiled, not run on hardware
//
#include <hip/hip_runtime.h>
#include <cstdint>
#include <cstddef>

typedef __attribute__((ext_vector_type(16))) __bf16 v16bf;
typedef __attribute__((ext_vector_type(8)))  __bf16 v8bf;
typedef __attribute__((ext_vector_type(8)))  float  v8f;
typedef __attribute__((ext_vector_type(4)))  int    v4i;

typedef __attribute__((address_space(1))) v4i* gptr_b128;
typedef __attribute__((address_space(3))) v4i* lptr_b128;

#define DEVINL static __device__ __forceinline__

constexpr int kB   = 8;
constexpr int kHW  = 64;
constexpr int kC   = 384;
constexpr int kF   = 1536;
constexpr int kE   = 8;
constexpr int kNP  = kB * 256;        // 2048 global patches
constexpr int kTok = kNP * 16;        // 32768 tokens
constexpr int kOutN = kB * kHW * kHW * kC;  // 12,582,912 floats
constexpr int kGW  = kC * 16;         // 6144 gate weights per expert

constexpr int kW1Stride = 392;        // LDS stride (elems) of w1 panel rows [32][C]
constexpr int kW2Stride = 40;         // LDS stride (elems) of w2 panel rows [C][32]
constexpr int kHStride  = 40;

// ---------------- fragment helpers (ISA 7.12.2 layouts) ----------------
DEVINL v16bf cat8(v8bf lo, v8bf hi) {
  return __builtin_shufflevector(lo, hi, 0,1,2,3,4,5,6,7,8,9,10,11,12,13,14,15);
}
// A-matrix 16-bit 16x32: lane's K runs are [half*8 .. +7] and [16+half*8 .. +7]
DEVINL v16bf frag_split(const __bf16* p) {
  v8bf lo = *reinterpret_cast<const v8bf*>(p);
  v8bf hi = *reinterpret_cast<const v8bf*>(p + 16);
  return cat8(lo, hi);
}
// A and B operands as loaded here (not ISA table order): lane-half holds 16 contiguous K values
DEVINL v16bf frag_cont(const __bf16* p) {
  v8bf lo = *reinterpret_cast<const v8bf*>(p);
  v8bf hi = *reinterpret_cast<const v8bf*>(p + 8);
  return cat8(lo, hi);
}
DEVINL v8f wmma_bf16(v16bf a, v16bf b, v8f c) {
  return __builtin_amdgcn_wmma_f32_16x16x32_bf16(false, a, false, b, (short)0, c,
                                                 false, false);
}
DEVINL float gelu_exact(float x) {
  return 0.5f * x * (1.0f + erff(x * 0.70710678118654752440f));
}

// 16-byte global->LDS async copy (ASYNCcnt path, GLOBAL_LOAD_ASYNC_TO_LDS_B128)
DEVINL void cp16_g2l(const __bf16* g, __bf16* l) {
  __builtin_amdgcn_global_load_async_to_lds_b128(
      (gptr_b128)const_cast<__bf16*>(g), (lptr_b128)l, 0, 0);
}
DEVINL void wait_async0() { __builtin_amdgcn_s_wait_asynccnt(0); }
// per-wave LDS store->load handoff (LDS ops complete in order within a wave)
DEVINL void lds_wave_fence() { asm volatile("s_wait_dscnt 0x0" ::: "memory"); }

// ---------------- setup kernels ----------------
__global__ void zero_out_kernel(float* __restrict__ out, int* __restrict__ cnt) {
  const int n4 = kOutN / 4;
  float4 z = make_float4(0.f, 0.f, 0.f, 0.f);
  float4* o4 = reinterpret_cast<float4*>(out);
  for (int i = blockIdx.x * blockDim.x + threadIdx.x; i < n4;
       i += gridDim.x * blockDim.x)
    o4[i] = z;
  if (blockIdx.x == 0 && threadIdx.x < kE) cnt[threadIdx.x] = 0;
}

// d[e][a][b] = (bf16) s[e][b][a]   (per-expert transpose + convert)
__global__ void transpose_cvt_kernel(const float* __restrict__ s,
                                     __bf16* __restrict__ d, int A, int Bdim) {
  const int per = A * Bdim;
  const int n = kE * per;
  for (int i = blockIdx.x * blockDim.x + threadIdx.x; i < n;
       i += gridDim.x * blockDim.x) {
    const int e = i / per;
    const int rem = i - e * per;
    const int a = rem / Bdim;
    const int b = rem - a * Bdim;
    d[i] = (__bf16)s[(size_t)e * per + (size_t)b * A + a];
  }
}

// Token-major gather of x with the reference's scramble folded in.
// Token t=(b,l2,p2) carries pixel (h,w)=((l2&15)*4+(l2>>6), p2*4+((l2>>4)&3)).
__global__ void gather_x_kernel(const float* __restrict__ x, __bf16* __restrict__ Xt) {
  const int n = kTok * kC;
  for (int i = blockIdx.x * blockDim.x + threadIdx.x; i < n;
       i += gridDim.x * blockDim.x) {
    const int c  = i % kC;
    const int t  = i / kC;
    const int p2 = t & 15;
    const int l2 = (t >> 4) & 255;
    const int b  = t >> 12;
    const int h  = (l2 & 15) * 4 + (l2 >> 6);
    const int w  = p2 * 4 + ((l2 >> 4) & 3);
    Xt[i] = (__bf16)x[((size_t)(b * (kHW * kHW) + h * kHW + w)) * kC + c];
  }
}

// ---------------- router: conv logits -> softmax -> top-2 -> expert lists ----
__global__ __launch_bounds__(128) void router_kernel(
    const float* __restrict__ x, const float* __restrict__ gw,
    int* __restrict__ cnt, int* __restrict__ plist, float* __restrict__ wlist) {
  __shared__ float xp[kGW];
  __shared__ float psum[kE][17];
  __shared__ float lg[kE];

  const int g  = blockIdx.x;  // global patch: b*256 + l2
  const int b  = g >> 8;
  const int l2 = g & 255;
  const int ph = l2 >> 4, pw = l2 & 15;
  const int tid = threadIdx.x;

  for (int q = 0; q < 16; ++q) {
    const int n = (ph * 4 + (q >> 2)) * kHW + pw * 4 + (q & 3);
    const float* src = x + ((size_t)(b * (kHW * kHW) + n)) * kC;
    for (int c = tid; c < kC; c += 128) xp[c * 16 + q] = src[c];
  }
  __syncthreads();

  {
    const int e = tid >> 4, s = tid & 15;
    const float* gwe = gw + e * kGW + s;
    float a = 0.f;
    for (int c = 0; c < kC; ++c) a += xp[c * 16 + s] * gwe[c * 16];
    psum[e][s] = a;
  }
  __syncthreads();
  if (tid < kE) {
    float a = 0.f;
#pragma unroll
    for (int s = 0; s < 16; ++s) a += psum[tid][s];
    lg[tid] = a;
  }
  __syncthreads();
  if (tid == 0) {
    float mx = lg[0];
#pragma unroll
    for (int e = 1; e < kE; ++e) mx = fmaxf(mx, lg[e]);
    float p[kE];
    float S = 0.f;
#pragma unroll
    for (int e = 0; e < kE; ++e) { p[e] = expf(lg[e] - mx); S += p[e]; }
#pragma unroll
    for (int e = 0; e < kE; ++e) p[e] /= S;
    int e1 = 0;
#pragma unroll
    for (int e = 1; e < kE; ++e) if (p[e] > p[e1]) e1 = e;
    int e2 = (e1 == 0) ? 1 : 0;
#pragma unroll
    for (int e = 0; e < kE; ++e) if (e != e1 && p[e] > p[e2]) e2 = e;
    const float s2 = p[e1] + p[e2] + 1e-9f;
    int pos = atomicAdd(&cnt[e1], 1);
    plist[e1 * 2048 + pos] = g;
    wlist[e1 * 2048 + pos] = p[e1] / s2;
    pos = atomicAdd(&cnt[e2], 1);
    plist[e2 * 2048 + pos] = g;
    wlist[e2 * 2048 + pos] = p[e2] / s2;
  }
}

// ---------------- fused expert MLP: gelu(X@w1+b1)@w2+b2, scatter-accumulate ----
// W1t is pre-transposed [E][F][C]; W2t is pre-transposed [E][C][F], both bf16,
// so both LDS panels are contiguous-row copies (async-LDS friendly).
__global__ __launch_bounds__(128, 1) void moe_expert_kernel(
    const __bf16* __restrict__ Xt, const __bf16* __restrict__ W1t,
    const __bf16* __restrict__ W2t, const float* __restrict__ b1,
    const float* __restrict__ b2, const int* __restrict__ cnt,
    const int* __restrict__ plist, const float* __restrict__ wlist,
    float* __restrict__ out) {
  __shared__ __align__(16) __bf16 w1buf[2][32 * kW1Stride];   // [f'][c]
  __shared__ __align__(16) __bf16 w2buf[2][kC * kW2Stride];   // [c][f']
  __shared__ __align__(16) __bf16 hbuf[4 * 16 * kHStride];    // per-wave H repack

  const int e    = blockIdx.x >> 9;
  const int tile = blockIdx.x & 511;
  const int cntE = cnt[e];
  const int base = tile * 4;
  if (base >= cntE) return;

  const int tid  = threadIdx.x;
  const int lane = tid & 31;
  const int wv   = tid >> 5;
  const int half = lane >> 4;
  const int ln   = lane & 15;

  const int  slot  = base + wv;
  const bool valid = slot < cntE;
  const int  sl    = valid ? slot : base;   // clamp to a real patch, weight 0
  const int  g     = plist[e * 2048 + sl];
  const float wgt  = valid ? wlist[e * 2048 + sl] : 0.0f;

  const __bf16* W1e = W1t + (size_t)e * kF * kC;   // [f][c]
  const __bf16* W2e = W2t + (size_t)e * kC * kF;   // [c][f]

  // X A-fragments for all 12 K-steps of GEMM1, kept in registers (96 VGPRs)
  v16bf xf[12];
  {
    const __bf16* xrow = Xt + (size_t)(g * 16 + ln) * kC + half * 16;
#pragma unroll
    for (int kk = 0; kk < 12; ++kk) xf[kk] = frag_cont(xrow + kk * 32);
  }

  v8f acc[24];
  v8f vzero = {};
#pragma unroll
  for (int i = 0; i < 24; ++i) acc[i] = vzero;

  __bf16* hrow = hbuf + wv * (16 * kHStride);
  const float* b1e = b1 + e * kF;

  // --- panel staging: contiguous 16B chunks via async LDS loads ---
  auto stage = [&](int buf, int f0) {
    __bf16* w1d = &w1buf[buf][0];
    __bf16* w2d = &w2buf[buf][0];
    // w1 panel: 32 rows x 384 elems  (row r = feature f0+r)
    for (int idx = tid; idx < 32 * 48; idx += 128) {
      const int r = idx / 48, c16 = idx - r * 48;
      cp16_g2l(W1e + (size_t)(f0 + r) * kC + c16 * 8,
               w1d + r * kW1Stride + c16 * 8);
    }
    // w2 panel: 384 rows x 32 elems (row c, cols f0..f0+31)
    for (int idx = tid; idx < kC * 4; idx += 128) {
      const int r = idx >> 2, c16 = idx & 3;
      cp16_g2l(W2e + (size_t)r * kF + f0 + c16 * 8,
               w2d + r * kW2Stride + c16 * 8);
    }
  };

  stage(0, 0);   // prologue: issue async loads for first panel

#pragma unroll 1
  for (int it = 0; it < kF / 32; ++it) {
    const int cur = it & 1;
    wait_async0();      // my async panel writes for buf[cur] are done
    __syncthreads();    // everyone's writes done; everyone done reading buf[cur^1]
    if (it + 1 < kF / 32) stage(cur ^ 1, (it + 1) * 32);  // prefetch next panel

    const int f0 = it * 32;
    const __bf16* w1p = &w1buf[cur][0];
    const __bf16* w2p = &w2buf[cur][0];

    // GEMM1: H[16x32] = X[16x384] @ w1[384 x (f0..f0+32)], depth-2 pipelined
    v8f h0 = vzero, h1 = vzero;
    v16bf bA = frag_cont(w1p + (0 + ln) * kW1Stride + half * 16);
    v16bf bB = frag_cont(w1p + (16 + ln) * kW1Stride + half * 16);
#pragma unroll
    for (int kk = 0; kk < 12; ++kk) {
      v16bf nA = bA, nB = bB;
      if (kk + 1 < 12) {
        nA = frag_cont(w1p + (0 + ln) * kW1Stride + (kk + 1) * 32 + half * 16);
        nB = frag_cont(w1p + (16 + ln) * kW1Stride + (kk + 1) * 32 + half * 16);
      }
      h0 = wmma_bf16(xf[kk], bA, h0);
      h1 = wmma_bf16(xf[kk], bB, h1);
      bA = nA; bB = nB;
    }

    // bias + exact GELU -> bf16, park in per-wave LDS for layout change
    const float ba = b1e[f0 + ln];
    const float bb = b1e[f0 + 16 + ln];
#pragma unroll
    for (int r = 0; r < 8; ++r) {
      const int m = r + 8 * half;              // C/D layout: row = r + 8*half
      hrow[m * kHStride + ln]      = (__bf16)gelu_exact(h0[r] + ba);
      hrow[m * kHStride + 16 + ln] = (__bf16)gelu_exact(h1[r] + bb);
    }
    lds_wave_fence();                          // wave-local store->load handoff

    // GEMM2: OUT[16x384] += H[16x32] @ w2[(f0..f0+32) x 384], depth-2 pipelined
    const v16bf af = frag_cont(hrow + ln * kHStride + half * 16);
    v16bf bc = frag_cont(w2p + ln * kW2Stride + half * 16);
#pragma unroll
    for (int nt = 0; nt < 24; ++nt) {
      v16bf bn = bc;
      if (nt + 1 < 24) {
        bn = frag_cont(w2p + ((nt + 1) * 16 + ln) * kW2Stride + half * 16);
      }
      acc[nt] = wmma_bf16(af, bc, acc[nt]);
      bc = bn;
    }
  }

  if (!valid) return;

  // epilogue: +b2, scale by routing weight, scatter-accumulate to permuted out
  const float* b2e = b2 + e * kC;
  const int b  = g >> 8;
  const int l2 = g & 255;
  const int hi = (l2 & 15) * 4 + (l2 >> 6);    // image row (fixed per patch)
  const int wb = (l2 >> 4) & 3;                // kw component of image col
  float* outb = out + (size_t)b * (kHW * kHW * kC) + (size_t)hi * (kC * kHW);
#pragma unroll
  for (int nt = 0; nt < 24; ++nt) {
    const int c = nt * 16 + ln;
    const float bias2 = b2e[c];
#pragma unroll
    for (int r = 0; r < 8; ++r) {
      const int p2 = r + 8 * half;             // token position within patch
      const int wi = p2 * 4 + wb;              // image col
      atomicAdd(outb + (size_t)c * kHW + wi, wgt * (acc[nt][r] + bias2));
    }
  }
}

// ---------------- host ----------------
extern "C" void kernel_launch(void* const* d_in, const int* in_sizes, int n_in,
                              void* d_out, int out_size, void* d_ws, size_t ws_size,
                              hipStream_t stream) {
  (void)in_sizes; (void)n_in; (void)out_size; (void)ws_size;
  const float* x  = (const float*)d_in[0];
  const float* gw = (const float*)d_in[1];
  const float* w1 = (const float*)d_in[2];
  const float* b1 = (const float*)d_in[3];
  const float* w2 = (const float*)d_in[4];
  const float* b2 = (const float*)d_in[5];
  float* out = (float*)d_out;

  char* ws = (char*)d_ws;                       // ~44.2 MB used
  __bf16* Xt  = (__bf16*)(ws);                  // 25,165,824 B  (token-major x)
  __bf16* W1t = (__bf16*)(ws + 25165824);       //  9,437,184 B  [E][F][C]
  __bf16* W2t = (__bf16*)(ws + 34603008);       //  9,437,184 B  [E][C][F]
  int*    cnt = (int*)   (ws + 44040192);       //  per-expert counters
  int*    pl  = (int*)   (ws + 44040448);       //  expert patch lists [E][2048]
  float*  wl  = (float*) (ws + 44105984);       //  expert patch weights

  zero_out_kernel<<<4096, 256, 0, stream>>>(out, cnt);
  router_kernel<<<kNP, 128, 0, stream>>>(x, gw, cnt, pl, wl);
  gather_x_kernel<<<8192, 256, 0, stream>>>(x, Xt);
  transpose_cvt_kernel<<<4096, 256, 0, stream>>>(w1, W1t, kF, kC);  // [E][F][C]
  transpose_cvt_kernel<<<4096, 256, 0, stream>>>(w2, W2t, kC, kF);  // [E][C][F]
  moe_expert_kernel<<<kE * 512, 128, 0, stream>>>(Xt, W1t, W2t, b1, b2, cnt, pl, wl,
                                                  out);
}
